// VQEmbedding_54374285967701
// MI455X (gfx1250) — matrix operand order repaired and checked by reading, not compiled, not run on hardware
//
#include <hip/hip_runtime.h>
#include <hip/hip_bf16.h>

typedef __attribute__((ext_vector_type(16))) __bf16 v16bf;
typedef __attribute__((ext_vector_type(8)))  float  v8f;

#define NUM_EMB   1024
#define EMB_DIM   64
#define N_ROWS    131072        // 32 * 64 * 64
#define HW        4096          // 64 * 64
#define BSTRIDE   262144        // EMB_DIM * HW
#define OUT_ELEMS 8388608       // 32 * 64 * 64 * 64
#define LOSS_SCALE (1.25f / 8388608.0f)

// ---------------------------------------------------------------------------
// Prep: codebook f32 -> bf16 (row-major, matches WMMA B-fragment contiguity),
// per-code squared norms, and zero the loss accumulator slot.
// grid = 1024 blocks (one per code), block = 32 threads (one wave).
// ---------------------------------------------------------------------------
__global__ void vq_prep(const float* __restrict__ W,
                        __bf16* __restrict__ wbf,
                        float* __restrict__ wsq,
                        float* __restrict__ loss) {
    const int k = blockIdx.x;          // code index
    const int t = threadIdx.x;         // 0..31
    const float w0 = W[k * EMB_DIM + t];
    const float w1 = W[k * EMB_DIM + t + 32];
    wbf[k * EMB_DIM + t]      = (__bf16)w0;
    wbf[k * EMB_DIM + t + 32] = (__bf16)w1;
    float s = w0 * w0 + w1 * w1;
    #pragma unroll
    for (int m = 16; m >= 1; m >>= 1) s += __shfl_xor(s, m, 32);
    if (t == 0) wsq[k] = s;
    if (k == 0 && t == 0) *loss = 0.0f;
}

// ---------------------------------------------------------------------------
// Main: per wave, 32 rows x 1024 codes via v_wmma_f32_16x16x32_bf16,
// running argmin in registers, xor-shuffle reduce, gather + loss + stores.
// grid = 256 blocks, block = 256 threads (8 waves), 2 passes per wave.
// Dynamic LDS: 128KB bf16 codebook + 4KB norms + per-wave index staging.
// ---------------------------------------------------------------------------
__global__ void __launch_bounds__(256, 1) vq_main(
    const float*  __restrict__ in,     // [32,64,64,64] NCHW
    const float*  __restrict__ W,      // [1024,64] f32
    const __bf16* __restrict__ wbf_g,  // [1024,64] bf16 (ws)
    const float*  __restrict__ wsq_g,  // [1024]    f32  (ws)
    float* __restrict__ out,           // [32,64,64,64] NCHW quantized
    float* __restrict__ loss,          // scalar accumulator
    float* __restrict__ idx_out)       // [131072] indices as float
{
    extern __shared__ char smem[];
    __bf16* wbf  = (__bf16*)smem;                        // 131072 B
    float*  wsq  = (float*)(smem + 131072);              // 4096 B
    int*    sidx = (int*)(smem + 131072 + 4096);         // 8 waves * 32 ints

    const int tid  = threadIdx.x;
    const int lane = tid & 31;
    const int wv   = tid >> 5;
    const int half = (lane >> 4) & 1;   // which 16-lane half
    const int l16  = lane & 15;

    // ---- stage codebook (bf16) + norms into LDS, shared by all 8 waves ----
    {
        const uint4* src = (const uint4*)wbf_g;
        uint4* dst = (uint4*)wbf;
        for (int i = tid; i < (NUM_EMB * EMB_DIM * 2) / 16; i += 256)
            dst[i] = src[i];
        for (int i = tid; i < NUM_EMB; i += 256)
            wsq[i] = wsq_g[i];
    }
    __syncthreads();

    float lsum = 0.0f;
    const int gw = blockIdx.x * 8 + wv;        // global wave id 0..2047

    for (int pass = 0; pass < 2; ++pass) {
        const int nb = (gw * 2 + pass) * 32;   // base flat row (multiple of 32)

        // ---- build A fragments: 2 row-tiles x 2 K-chunks (f32 -> bf16) ----
        // A 16x32 layout (matches B): lanes 0-15 row=l16 hold K 0..15,
        // lanes 16-31 row=l16 hold K 16..31.
        v16bf a[2][2];
        #pragma unroll
        for (int rt = 0; rt < 2; ++rt) {
            const int n  = nb + rt * 16 + l16;
            const int b  = n >> 12;            // batch (HW = 4096 rows/batch)
            const int hw = n & (HW - 1);
            const float* xp = in + (size_t)b * BSTRIDE + hw;  // stride HW per d
            #pragma unroll
            for (int c = 0; c < 2; ++c) {
                const int dbase = c * 32 + half * 16;
                #pragma unroll
                for (int i = 0; i < 16; ++i) {
                    const int d = dbase + i;
                    a[rt][c][i] = (__bf16)xp[(size_t)d * HW];
                }
            }
        }

        // ---- running argmin over 64 code-tiles ----
        float best[2][8];
        int   bestk[2][8];
        #pragma unroll
        for (int rt = 0; rt < 2; ++rt)
            #pragma unroll
            for (int j = 0; j < 8; ++j) { best[rt][j] = 3.4e38f; bestk[rt][j] = 0; }

        for (int t = 0; t < 64; ++t) {
            const int kcol = t * 16 + l16;     // this lane's code column
            // B Kx16 layout used here (contiguous K, same as A): lane holds 16 contiguous K (dims) of
            // column kcol; lanes 0-15 get K 0..15, lanes 16-31 get K 16..31.
            const v16bf b0 = *(const v16bf*)(wbf + kcol * EMB_DIM + half * 16);
            const v16bf b1 = *(const v16bf*)(wbf + kcol * EMB_DIM + 32 + half * 16);
            const float wsqk = wsq[kcol];
            #pragma unroll
            for (int rt = 0; rt < 2; ++rt) {
                v8f c = {};
                c = __builtin_amdgcn_wmma_f32_16x16x32_bf16(
                        false, a[rt][0], false, b0, (short)0, c, false, false);
                c = __builtin_amdgcn_wmma_f32_16x16x32_bf16(
                        false, a[rt][1], false, b1, (short)0, c, false, false);
                // C layout: VGPR j -> row j (lanes<16) / j+8 (lanes>=16),
                // column = kcol.  score = ||w||^2 - 2*dot  (||x||^2 constant)
                #pragma unroll
                for (int j = 0; j < 8; ++j) {
                    const float sc = fmaf(-2.0f, c[j], wsqk);
                    if (sc < best[rt][j]) { best[rt][j] = sc; bestk[rt][j] = kcol; }
                }
            }
        }

        // ---- reduce argmin across the 16 lanes sharing each row group ----
        #pragma unroll
        for (int rt = 0; rt < 2; ++rt) {
            #pragma unroll
            for (int j = 0; j < 8; ++j) {
                float bs = best[rt][j];
                int   bk = bestk[rt][j];
                #pragma unroll
                for (int m = 1; m <= 8; m <<= 1) {
                    const float os = __shfl_xor(bs, m, 32);
                    const int   ok = __shfl_xor(bk, m, 32);
                    if (os < bs || (os == bs && ok < bk)) { bs = os; bk = ok; }
                }
                if (l16 == 0)   // lanes 0 (rows 0..7) and 16 (rows 8..15)
                    sidx[wv * 32 + rt * 16 + half * 8 + j] = bk;
            }
        }
        __syncthreads();

        // ---- gather w[idx], quantized store (coalesced over hw), loss ----
        for (int i = lane; i < 32 * EMB_DIM; i += 32) {
            const int r   = i & 31;
            const int d   = i >> 5;
            const int idx = sidx[wv * 32 + r];
            const int n   = nb + r;
            const int b   = n >> 12;
            const int hw  = n & (HW - 1);
            const size_t addr = (size_t)b * BSTRIDE + (size_t)d * HW + hw;
            const float xv = in[addr];
            const float qv = W[idx * EMB_DIM + d];
            const float df = qv - xv;
            lsum = fmaf(df, df, lsum);
            out[addr] = qv;   // straight-through forward == quantized
        }
        idx_out[nb + lane] = (float)sidx[wv * 32 + lane];
    }

    // ---- loss: wave reduce, one atomic per wave ----
    #pragma unroll
    for (int m = 16; m >= 1; m >>= 1) lsum += __shfl_xor(lsum, m, 32);
    if (lane == 0) atomicAdd(loss, lsum * LOSS_SCALE);
}

// ---------------------------------------------------------------------------
extern "C" void kernel_launch(void* const* d_in, const int* in_sizes, int n_in,
                              void* d_out, int out_size, void* d_ws, size_t ws_size,
                              hipStream_t stream) {
    const float* in = (const float*)d_in[0];   // [32,64,64,64] f32
    const float* W  = (const float*)d_in[1];   // [1024,64] f32

    float* out  = (float*)d_out;               // 8388608 quantized
    float* loss = out + OUT_ELEMS;             // +1 scalar loss
    float* idx  = out + OUT_ELEMS + 1;         // +131072 indices (as float)

    __bf16* wbf = (__bf16*)d_ws;                       // 128 KB
    float*  wsq = (float*)((char*)d_ws + 131072);      // 4 KB

    vq_prep<<<NUM_EMB, 32, 0, stream>>>(W, wbf, wsq, loss);

    const size_t shmem = 131072 + 4096 + 8 * 32 * sizeof(int);
    vq_main<<<256, 256, shmem, stream>>>(in, W, wbf, wsq, out, loss, idx);
}
